// SceneMapEncoderDecoder_21758304322193
// MI455X (gfx1250) — matrix operand order repaired and checked by reading, not compiled, not run on hardware
//
#include <hip/hip_runtime.h>
#include <hip/hip_bf16.h>
#include <stdint.h>

#define BN_SC 0.99999500003749975f
#define CEILDIV(a,b) (((a)+(b)-1)/(b))

typedef __attribute__((ext_vector_type(16))) __bf16    v16bf;
typedef __attribute__((ext_vector_type(8)))  float     v8f;
typedef __attribute__((ext_vector_type(8)))  unsigned  v8u;

static __device__ inline unsigned short bf16b(float f) {
  unsigned u = __builtin_bit_cast(unsigned, f);
  unsigned r = u + 0x7FFFu + ((u >> 16) & 1u);   // round-to-nearest-even
  return (unsigned short)(r >> 16);
}
static __device__ inline unsigned pk2bf(float lo, float hi) {
  return ((unsigned)bf16b(hi) << 16) | (unsigned)bf16b(lo);
}

// ---------------------------------------------------------------------------
// WMMA GEMM: C[M,N] = epi(A[M,K] @ W[K,N]); epi = +bias -> BN -> +res -> ReLU
// One wave per 16x32 C tile (two 16x16 WMMA accumulators sharing A fragment).
// Tiles staged through LDS as packed bf16 in the exact ISA fragment layout:
//   A: lA[row][kpair]  (16x32 A-matrix, 16-bit, per ISA 7.12.2)
//   B: lB[col][kpair]  (32x32 -> two 32x16 B operands)
// Interior tiles take a wave-uniform fast path with fully unguarded loads;
// edge tiles use index-clamped + cndmask-zeroed loads (no per-lane branches).
// ---------------------------------------------------------------------------
__global__ void __launch_bounds__(32) k_gemm(
    const float* __restrict__ A, const float* __restrict__ W,
    const float* __restrict__ bias, const float* __restrict__ gam,
    const float* __restrict__ bet, const float* __restrict__ res,
    int relu, float* __restrict__ C, int M, int K, int N)
{
  __shared__ unsigned lA[16][16];   // [row][kpair] packed 2x bf16
  __shared__ unsigned lB[32][16];   // [col][kpair] packed 2x bf16
  const int lane = threadIdx.x;
  const int half = lane >> 4;
  const int lr   = lane & 15;
  const int row0 = blockIdx.x << 4;
  const int col0 = blockIdx.y << 5;
  v8f acc0 = {0.f,0.f,0.f,0.f,0.f,0.f,0.f,0.f};
  v8f acc1 = {0.f,0.f,0.f,0.f,0.f,0.f,0.f,0.f};
  // per-lane fixed (row,kpair) assignments for staging
  const int ar  = lane >> 1;            // A row handled by this lane (2 lanes/row)
  const int ak0 = (lane & 1) * 8;       // first of 8 kpairs for this lane
  for (int kb = 0; kb < K; kb += 32) {
    // ---- stage A tile: 16 rows x 16 kpairs, 8 dwords per lane -------------
    if (row0 + 16 <= M && kb + 32 <= K) {        // wave-uniform fast path
      const float* ap = A + (size_t)(row0 + ar) * K + kb + 2*ak0;
#pragma unroll
      for (int t = 0; t < 8; ++t)
        lA[ar][ak0 + t] = pk2bf(ap[2*t], ap[2*t + 1]);
    } else {
#pragma unroll
      for (int t = 0; t < 8; ++t) {
        int kp = ak0 + t;
        int gr = row0 + ar;
        int k0 = kb + 2*kp, k1 = k0 + 1;
        int rc  = gr < M ? gr : M - 1;
        int kc0 = k0 < K ? k0 : K - 1;
        int kc1 = k1 < K ? k1 : K - 1;
        float a0 = A[(size_t)rc * K + kc0];
        float a1 = A[(size_t)rc * K + kc1];
        a0 = (gr < M && k0 < K) ? a0 : 0.0f;
        a1 = (gr < M && k1 < K) ? a1 : 0.0f;
        lA[ar][kp] = pk2bf(a0, a1);
      }
    }
    // ---- stage B tile: 32 cols x 16 kpairs, 16 dwords per lane ------------
    if (col0 + 32 <= N && kb + 32 <= K) {        // wave-uniform fast path
      const float* wp = W + (size_t)kb * N + col0 + lane;
#pragma unroll
      for (int kp = 0; kp < 16; ++kp)
        lB[lane][kp] = pk2bf(wp[(size_t)(2*kp) * N], wp[(size_t)(2*kp + 1) * N]);
    } else {
#pragma unroll
      for (int kp = 0; kp < 16; ++kp) {
        int gc = col0 + lane;
        int k0 = kb + 2*kp, k1 = k0 + 1;
        int cc  = gc < N ? gc : N - 1;
        int kc0 = k0 < K ? k0 : K - 1;
        int kc1 = k1 < K ? k1 : K - 1;
        float b0 = W[(size_t)kc0 * N + cc];
        float b1 = W[(size_t)kc1 * N + cc];
        b0 = (gc < N && k0 < K) ? b0 : 0.0f;
        b1 = (gc < N && k1 < K) ? b1 : 0.0f;
        lB[lane][kp] = pk2bf(b0, b1);
      }
    }
    __syncthreads();
    // ---- fragment reads: pure ds_load_b128, no VALU reshuffle -------------
    v8u au, bu0, bu1;
#pragma unroll
    for (int d = 0; d < 8; ++d) {
      int kpa = ((d < 4) ? d : 4 + d) + 4 * half;  // A 16-bit 16x32 layout
      int kpb = ((d < 4) ? d : 4 + d) + 4 * half;  // B 32x16, same k order as A
      au[d]  = lA[lr][kpa];
      bu0[d] = lB[lr][kpb];
      bu1[d] = lB[16 + lr][kpb];
    }
    v16bf af  = __builtin_bit_cast(v16bf, au);
    v16bf bf0 = __builtin_bit_cast(v16bf, bu0);
    v16bf bf1 = __builtin_bit_cast(v16bf, bu1);
    acc0 = __builtin_amdgcn_wmma_f32_16x16x32_bf16(false, af, false, bf0,
                                                   (short)0, acc0, false, false);
    acc1 = __builtin_amdgcn_wmma_f32_16x16x32_bf16(false, af, false, bf1,
                                                   (short)0, acc1, false, false);
    __syncthreads();
  }
  // ---- epilogue ------------------------------------------------------------
#pragma unroll
  for (int v = 0; v < 8; ++v) {
    const int m = row0 + v + 8 * half;
    if (m < M) {
#pragma unroll
      for (int h = 0; h < 2; ++h) {
        const int n = col0 + 16 * h + lr;
        if (n < N) {
          float val = h ? acc1[v] : acc0[v];
          if (bias) val += bias[n];
          if (gam)  val = gam[n] * (val * BN_SC) + bet[n];
          if (res)  val += res[(size_t)m * N + n];
          if (relu) val = fmaxf(val, 0.0f);
          C[(size_t)m * N + n] = val;
        }
      }
    }
  }
}

// ------------------------------ elementwise --------------------------------
__global__ void k_concat2(const float* a, int ca, const float* b, int cb,
                          float* o, int total) {
  int t = blockIdx.x * blockDim.x + threadIdx.x; if (t >= total) return;
  int cc = ca + cb; int ch = t % cc; int r = t / cc;
  o[t] = (ch < ca) ? a[(size_t)r*ca + ch] : b[(size_t)r*cb + (ch - ca)];
}

__global__ void k_catb(const float* x, int n, int c, const float* gvec,
                       float* o, int total) {  // concat(x, broadcast-per-batch g)
  int t = blockIdx.x * blockDim.x + threadIdx.x; if (t >= total) return;
  int cc = 2*c; int ch = t % cc; int r = t / cc; int b = r / n;
  o[t] = (ch < c) ? x[(size_t)r*c + ch] : gvec[(size_t)b*c + (ch - c)];
}

__global__ void k_bnrelu(float* x, const float* g, const float* be,
                         int total, int C) {
  int t = blockIdx.x * blockDim.x + threadIdx.x; if (t >= total) return;
  int ch = t % C;
  float v = g[ch] * (x[t] * BN_SC) + be[ch];
  x[t] = fmaxf(v, 0.0f);
}

__global__ void k_rel(const float* pos, const int* idx, int n, int ns,
                      float* rel, int total) {  // total = B*n*ns
  int t = blockIdx.x * blockDim.x + threadIdx.x; if (t >= total) return;
  int j = t % ns; int bi = t / ns; int b = bi / n;
  int tg = idx[(size_t)bi*ns + j];
  const float* pc = pos + (size_t)bi * 3;
  const float* pt = pos + ((size_t)b*n + tg) * 3;
  rel[(size_t)t*3 + 0] = pt[0] - pc[0];
  rel[(size_t)t*3 + 1] = pt[1] - pc[1];
  rel[(size_t)t*3 + 2] = pt[2] - pc[2];
}

__global__ void k_wcomb(const float* kf, const float* qf, const float* pr,
                        const int* idx, int n, int ns, int C,
                        float* o, int total) {  // total = B*n*ns*C
  int t = blockIdx.x * blockDim.x + threadIdx.x; if (t >= total) return;
  int ch = t % C; int r = t / C; int j = r % ns; int bi = r / ns; int b = bi / n;
  int tg = idx[(size_t)bi*ns + j];
  o[t] = kf[((size_t)b*n + tg)*C + ch] - qf[(size_t)bi*C + ch] + pr[t];
}

__global__ void k_softmax_ns(float* w, int rows, int ns, int cs, int total) {
  int t = blockIdx.x * blockDim.x + threadIdx.x; if (t >= total) return;
  int ch = t % cs; int r = t / cs;
  size_t base = (size_t)r * ns * cs + ch;
  float mx = -3.4e38f;
  for (int j = 0; j < ns; ++j) mx = fmaxf(mx, w[base + (size_t)j*cs]);
  float s = 0.f;
  for (int j = 0; j < ns; ++j) { float e = expf(w[base + (size_t)j*cs] - mx);
                                 w[base + (size_t)j*cs] = e; s += e; }
  float inv = 1.0f / s;
  for (int j = 0; j < ns; ++j) w[base + (size_t)j*cs] *= inv;
}

__global__ void k_attnout(const float* vf, const float* pr, const float* w,
                          const int* idx, int n, int ns, int C, int cs,
                          float* o, int total) {  // total = B*n*C
  int t = blockIdx.x * blockDim.x + threadIdx.x; if (t >= total) return;
  int ch = t % C; int bi = t / C; int b = bi / n;
  int cw = ch % cs;
  float acc = 0.f;
  for (int j = 0; j < ns; ++j) {
    int tg = idx[(size_t)bi*ns + j];
    size_t nj = (size_t)bi*ns + j;
    acc += (vf[((size_t)b*n + tg)*C + ch] + pr[nj*C + ch]) * w[nj*cs + cw];
  }
  o[t] = acc;
}

__global__ void k_tdg(const float* posp, const float* posn, const float* featp,
                      const int* idx, int m, int np, int ns, int cin,
                      float* g, int total) {  // total = B*m*ns*(3+cin)
  int t = blockIdx.x * blockDim.x + threadIdx.x; if (t >= total) return;
  int cg = 3 + cin; int ch = t % cg; int r = t / cg;
  int j = r % ns; int bs = r / ns; int b = bs / m;
  int tg = idx[(size_t)bs*ns + j];
  if (ch < 3) g[t] = posp[((size_t)b*np + tg)*3 + ch] - posn[(size_t)bs*3 + ch];
  else        g[t] = featp[((size_t)b*np + tg)*cin + (ch - 3)];
}

__global__ void k_maxns(const float* y, int ns, int C, float* o, int total) {
  int t = blockIdx.x * blockDim.x + threadIdx.x; if (t >= total) return;
  int ch = t % C; int bs = t / C;
  float mx = -3.4e38f;
  for (int j = 0; j < ns; ++j)
    mx = fmaxf(mx, y[((size_t)bs*ns + j)*C + ch]);
  o[t] = mx;
}

__global__ void k_gatherpos(const float* pos, const int* sidx, int np, int m,
                            float* o, int total) {  // total = B*m*3
  int t = blockIdx.x * blockDim.x + threadIdx.x; if (t >= total) return;
  int k = t % 3; int bs = t / 3; int b = bs / m; int s = bs % m;
  int tg = sidx[(size_t)b*m + s];
  o[t] = pos[((size_t)b*np + tg)*3 + k];
}

__global__ void k_mean(const float* x, int n, int C, float* o, int total) {
  int t = blockIdx.x * blockDim.x + threadIdx.x; if (t >= total) return;
  int ch = t % C; int b = t / C;
  float a = 0.f;
  for (int i = 0; i < n; ++i) a += x[((size_t)b*n + i)*C + ch];
  o[t] = a / (float)n;
}

__global__ void k_interp(const float* y1, const float* y2, const int* idx,
                         const float* d, int n, int m, int C,
                         float* o, int total) {  // total = B*n*C
  int t = blockIdx.x * blockDim.x + threadIdx.x; if (t >= total) return;
  int ch = t % C; int bi = t / C; int b = bi / n;
  float w0 = 1.0f / (d[(size_t)bi*3 + 0] + 1e-8f);
  float w1 = 1.0f / (d[(size_t)bi*3 + 1] + 1e-8f);
  float w2 = 1.0f / (d[(size_t)bi*3 + 2] + 1e-8f);
  float s = w0 + w1 + w2;
  int i0 = idx[(size_t)bi*3 + 0], i1 = idx[(size_t)bi*3 + 1], i2 = idx[(size_t)bi*3 + 2];
  float acc = w0 * y2[((size_t)b*m + i0)*C + ch]
            + w1 * y2[((size_t)b*m + i1)*C + ch]
            + w2 * y2[((size_t)b*m + i2)*C + ch];
  o[t] = y1[t] + acc / s;
}

// ------------------------------ FPS ----------------------------------------
__global__ void k_fps(const float* pos, int n, int m, int* sidx, float* dist) {
  int b = blockIdx.x;
  pos  += (size_t)b * n * 3;
  sidx += (size_t)b * m;
  dist += (size_t)b * n;
  int tid = threadIdx.x; int bd = blockDim.x;
  __shared__ float sv[256];
  __shared__ int   si[256];
  __shared__ int   slast;
  for (int i = tid; i < n; i += bd) dist[i] = 1e10f;
  if (tid == 0) { sidx[0] = 0; slast = 0; }
  __syncthreads();
  for (int s = 1; s < m; ++s) {
    int last = slast;
    float lx = pos[(size_t)last*3 + 0];
    float ly = pos[(size_t)last*3 + 1];
    float lz = pos[(size_t)last*3 + 2];
    float bmax = -1.0f; int bidx = 0x7fffffff;
    for (int i = tid; i < n; i += bd) {
      float dx = pos[(size_t)i*3+0] - lx;
      float dy = pos[(size_t)i*3+1] - ly;
      float dz = pos[(size_t)i*3+2] - lz;
      float dd = dx*dx + dy*dy + dz*dz;
      float nd = fminf(dist[i], dd);
      dist[i] = nd;
      if (nd > bmax) { bmax = nd; bidx = i; }
    }
    sv[tid] = bmax; si[tid] = bidx;
    __syncthreads();
    for (int off = 128; off > 0; off >>= 1) {
      if (tid < off) {
        if (sv[tid+off] > sv[tid] ||
            (sv[tid+off] == sv[tid] && si[tid+off] < si[tid])) {
          sv[tid] = sv[tid+off]; si[tid] = si[tid+off];
        }
      }
      __syncthreads();
    }
    if (tid == 0) { sidx[s] = si[0]; slast = si[0]; }
    __syncthreads();
  }
}

// ------------------------------ kNN ----------------------------------------
template<int K>
__global__ void k_knn(const float* q, int nq, const float* p, int np,
                      int* oi, float* od) {
  int b = blockIdx.y;
  int i = blockIdx.x * blockDim.x + threadIdx.x;
  if (i >= nq) return;
  const float* qp = q + ((size_t)b*nq + i) * 3;
  const float* pp = p + (size_t)b * np * 3;
  float qx = qp[0], qy = qp[1], qz = qp[2];
  float bd[K]; int bi[K];
#pragma unroll
  for (int t = 0; t < K; ++t) { bd[t] = 3.4e38f; bi[t] = 0; }
  for (int j = 0; j < np; ++j) {
    if ((j & 63) == 0 && j + 64 < np)
      __builtin_prefetch(&pp[(size_t)(j + 64) * 3], 0, 3);  // global_prefetch_b8
    float dx = pp[(size_t)j*3+0] - qx;
    float dy = pp[(size_t)j*3+1] - qy;
    float dz = pp[(size_t)j*3+2] - qz;
    float dd = dx*dx + dy*dy + dz*dz;
    if (dd < bd[K-1]) {
      int pos = K - 1;
#pragma unroll
      for (int t = K - 2; t >= 0; --t)
        if (dd < bd[t]) { bd[t+1] = bd[t]; bi[t+1] = bi[t]; pos = t; }
      bd[pos] = dd; bi[pos] = j;
    }
  }
  int* wi = oi + ((size_t)b*nq + i) * K;
#pragma unroll
  for (int t = 0; t < K; ++t) wi[t] = bi[t];
  if (od) {
    float* wd = od + ((size_t)b*nq + i) * K;
#pragma unroll
    for (int t = 0; t < K; ++t) wd[t] = bd[t];
  }
}

// ======================= host-side orchestration ===========================
struct Lin { const float* w; const float* b; };
struct Bn  { const float* g; const float* be; };
struct Ptl { Lin q, k, v, p1, p2, w1, w2; Bn pbn, wbn1, wbn2; };
struct Blk { Lin l1, l3; Bn bn1, bn2, bn3; Ptl tr; };
struct Cur { void* const* d; int i; const float* nx() { return (const float*)d[i++]; } };

static Bn  takeBn(Cur& c)            { Bn b; b.be = c.nx(); b.g = c.nx(); return b; }
static Lin takeLin(Cur& c, bool bias){ Lin l; l.b = bias ? c.nx() : nullptr; l.w = c.nx(); return l; }
static Ptl takePtl(Cur& c) {
  Ptl t;
  t.k  = takeLin(c, true);  t.p1 = takeLin(c, true);  t.p2 = takeLin(c, true);
  t.pbn = takeBn(c);
  t.q  = takeLin(c, true);  t.v  = takeLin(c, true);
  t.w1 = takeLin(c, true);  t.w2 = takeLin(c, true);
  t.wbn1 = takeBn(c);       t.wbn2 = takeBn(c);
  return t;
}
static Blk takeBlk(Cur& c) {
  Blk b;
  b.bn1 = takeBn(c); b.bn2 = takeBn(c); b.bn3 = takeBn(c);
  b.l1 = takeLin(c, false); b.l3 = takeLin(c, false);
  b.tr = takePtl(c);
  return b;
}

struct DecP { Blk blk; Bn bn1, bn2; Lin l1, l2; };
struct EncP { Blk blocks[5]; int nb; Bn td_bn; Lin td_lin; };
struct Bufs { float *qb,*kb,*vb,*ty,*ty2,*rel,*tmp3,*pr,*wb,*wb2,*wb3; };

static void gemm(hipStream_t s, const float* A, const float* W, const float* bias,
                 const float* g, const float* be, const float* res, int relu,
                 float* C, int M, int K, int N) {
  k_gemm<<<dim3(CEILDIV(M,16), CEILDIV(N,32)), dim3(32), 0, s>>>(
      A, W, bias, g, be, res, relu, C, M, K, N);
}

#define LAUNCH1D(kern, total, ...) \
  kern<<<dim3((unsigned)CEILDIV((total),256)), dim3(256), 0, s>>>(__VA_ARGS__)

static void run_block(hipStream_t s, const float* pos, float* feat, int n, int c,
                      int ns, const int* idx, const Blk& bk, const Bufs& bf) {
  const int M  = 2 * n;
  const int Mn = M * ns;
  const int cs = c / 8;
  // y = relu(bn1(l1(x)))
  gemm(s, feat, bk.l1.w, nullptr, bk.bn1.g, bk.bn1.be, nullptr, 1, bf.ty, M, c, c);
  // q,k,v
  gemm(s, bf.ty, bk.tr.q.w, bk.tr.q.b, nullptr, nullptr, nullptr, 0, bf.qb, M, c, c);
  gemm(s, bf.ty, bk.tr.k.w, bk.tr.k.b, nullptr, nullptr, nullptr, 0, bf.kb, M, c, c);
  gemm(s, bf.ty, bk.tr.v.w, bk.tr.v.b, nullptr, nullptr, nullptr, 0, bf.vb, M, c, c);
  // rel pos, positional MLP p_r
  LAUNCH1D(k_rel, Mn, pos, idx, n, ns, bf.rel, Mn);
  gemm(s, bf.rel,  bk.tr.p1.w, bk.tr.p1.b, bk.tr.pbn.g, bk.tr.pbn.be, nullptr, 1,
       bf.tmp3, Mn, 3, 3);
  gemm(s, bf.tmp3, bk.tr.p2.w, bk.tr.p2.b, nullptr, nullptr, nullptr, 0,
       bf.pr, Mn, 3, c);
  // w = gather(k) - q + p_r ; two-layer MLP with BN+ReLU; softmax over ns
  LAUNCH1D(k_wcomb, Mn*c, bf.kb, bf.qb, bf.pr, idx, n, ns, c, bf.wb, Mn*c);
  LAUNCH1D(k_bnrelu, Mn*c, bf.wb, bk.tr.wbn1.g, bk.tr.wbn1.be, Mn*c, c);
  gemm(s, bf.wb,  bk.tr.w1.w, bk.tr.w1.b, nullptr, nullptr, nullptr, 0, bf.wb2, Mn, c, cs);
  LAUNCH1D(k_bnrelu, Mn*cs, bf.wb2, bk.tr.wbn2.g, bk.tr.wbn2.be, Mn*cs, cs);
  gemm(s, bf.wb2, bk.tr.w2.w, bk.tr.w2.b, nullptr, nullptr, nullptr, 0, bf.wb3, Mn, cs, cs);
  LAUNCH1D(k_softmax_ns, M*cs, bf.wb3, M, ns, cs, M*cs);
  // out = sum_j (v[idx]+p_r) * w  (grouped, share=8)
  LAUNCH1D(k_attnout, M*c, bf.vb, bf.pr, bf.wb3, idx, n, ns, c, cs, bf.ty2, M*c);
  // relu(bn2(.)) then relu(bn3(l3(.)) + x)
  LAUNCH1D(k_bnrelu, M*c, bf.ty2, bk.bn2.g, bk.bn2.be, M*c, c);
  gemm(s, bf.ty2, bk.l3.w, nullptr, bk.bn3.g, bk.bn3.be, feat, 1, feat, M, c, c);
}

extern "C" void kernel_launch(void* const* d_in, const int* in_sizes, int n_in,
                              void* d_out, int out_size, void* d_ws, size_t ws_size,
                              hipStream_t stream) {
  (void)in_sizes; (void)n_in; (void)out_size; (void)ws_size;
  hipStream_t s = stream;
  const float* P = (const float*)d_in[0];   // (B,8192,3)
  const float* X = (const float*)d_in[1];   // (B,8192,3)

  // ---- parameter binding (jax pytree flatten: dict keys sorted, lists in order)
  Cur cur{d_in, 2};
  DecP dec[4];
  for (int i = 0; i < 4; ++i) {            // dec1, dec2, dec3, dec4
    dec[i].blk = takeBlk(cur);             // 'blocks' < 'tu'
    dec[i].bn1 = takeBn(cur);
    if (i != 3) dec[i].bn2 = takeBn(cur);  // dec4 tu has no bn2
    else        dec[i].bn2 = Bn{nullptr, nullptr};
    dec[i].l1 = takeLin(cur, true);
    dec[i].l2 = takeLin(cur, true);
  }
  EncP enc[4];
  const int nbk[4] = {1, 2, 3, 5};
  for (int i = 0; i < 4; ++i) {            // enc1..enc4
    enc[i].nb = nbk[i];
    for (int j = 0; j < nbk[i]; ++j) enc[i].blocks[j] = takeBlk(cur);
    enc[i].td_bn  = takeBn(cur);           // td: 'bn' < 'lin'
    enc[i].td_lin = takeLin(cur, false);
  }

  // ---- workspace carving
  float* base = (float*)d_ws;
  size_t off = 0;
  auto AL = [&](size_t cnt) -> float* {
    float* r = base + off; off += (cnt + 63) & ~(size_t)63; return r; };
  float* feat0 = AL(2*8192*6);
  float* pos2  = AL(2*2048*3);
  float* pos3  = AL(2*512*3);
  float* pos4  = AL(2*128*3);
  float* feat1 = AL(2*8192*32);
  float* feat2 = AL(2*2048*64);
  float* feat3 = AL(2*512*128);
  float* feat4 = AL(2*128*256);
  int* idx1  = (int*)AL(2*8192*8);
  int* idx2  = (int*)AL(2*2048*16);
  int* idx3  = (int*)AL(2*512*16);
  int* idx4  = (int*)AL(2*128*16);
  float* fpsd = AL(2*8192);
  int* sidx  = (int*)AL(2*2048);
  int* tdidx = (int*)AL(2*2048*16);
  int* iidx  = (int*)AL(2*8192*3);
  float* idist = AL(2*8192*3);
  Bufs bf;
  bf.qb = AL(524288); bf.kb = AL(524288); bf.vb = AL(524288);
  bf.ty = AL(524288); bf.ty2 = AL(524288);
  bf.rel = AL(393216); bf.tmp3 = AL(393216);
  bf.pr = AL(4194304); bf.wb = AL(4194304);
  bf.wb2 = AL(524288); bf.wb3 = AL(524288);
  float* gbuf = AL(2293760);
  float* ybuf = AL(4194304);
  float* meanb = AL(512);
  float* gvec  = AL(512);

  // ======================== ENCODER ========================
  // feat0 = cat(p, x)
  LAUNCH1D(k_concat2, 2*8192*6, P, 3, X, 3, feat0, 2*8192*6);
  // enc1 td (stride 1): relu(bn(lin 6->32))
  gemm(s, feat0, enc[0].td_lin.w, nullptr, enc[0].td_bn.g, enc[0].td_bn.be,
       nullptr, 1, feat1, 2*8192, 6, 32);
  k_knn<8><<<dim3(CEILDIV(8192,128), 2), dim3(128), 0, s>>>(P, 8192, P, 8192, idx1, nullptr);
  for (int j = 0; j < enc[0].nb; ++j)
    run_block(s, P, feat1, 8192, 32, 8, idx1, enc[0].blocks[j], bf);

  // enc2 td (stride 4): 8192 -> 2048, din=35 -> 64
  k_fps<<<dim3(2), dim3(256), 0, s>>>(P, 8192, 2048, sidx, fpsd);
  LAUNCH1D(k_gatherpos, 2*2048*3, P, sidx, 8192, 2048, pos2, 2*2048*3);
  k_knn<16><<<dim3(CEILDIV(2048,128), 2), dim3(128), 0, s>>>(pos2, 2048, P, 8192, tdidx, nullptr);
  LAUNCH1D(k_tdg, 2*2048*16*35, P, pos2, feat1, tdidx, 2048, 8192, 16, 32, gbuf, 2*2048*16*35);
  gemm(s, gbuf, enc[1].td_lin.w, nullptr, enc[1].td_bn.g, enc[1].td_bn.be,
       nullptr, 1, ybuf, 2*2048*16, 35, 64);
  LAUNCH1D(k_maxns, 2*2048*64, ybuf, 16, 64, feat2, 2*2048*64);
  k_knn<16><<<dim3(CEILDIV(2048,128), 2), dim3(128), 0, s>>>(pos2, 2048, pos2, 2048, idx2, nullptr);
  for (int j = 0; j < enc[1].nb; ++j)
    run_block(s, pos2, feat2, 2048, 64, 16, idx2, enc[1].blocks[j], bf);

  // enc3 td: 2048 -> 512, din=67 -> 128
  k_fps<<<dim3(2), dim3(256), 0, s>>>(pos2, 2048, 512, sidx, fpsd);
  LAUNCH1D(k_gatherpos, 2*512*3, pos2, sidx, 2048, 512, pos3, 2*512*3);
  k_knn<16><<<dim3(CEILDIV(512,128), 2), dim3(128), 0, s>>>(pos3, 512, pos2, 2048, tdidx, nullptr);
  LAUNCH1D(k_tdg, 2*512*16*67, pos2, pos3, feat2, tdidx, 512, 2048, 16, 64, gbuf, 2*512*16*67);
  gemm(s, gbuf, enc[2].td_lin.w, nullptr, enc[2].td_bn.g, enc[2].td_bn.be,
       nullptr, 1, ybuf, 2*512*16, 67, 128);
  LAUNCH1D(k_maxns, 2*512*128, ybuf, 16, 128, feat3, 2*512*128);
  k_knn<16><<<dim3(CEILDIV(512,128), 2), dim3(128), 0, s>>>(pos3, 512, pos3, 512, idx3, nullptr);
  for (int j = 0; j < enc[2].nb; ++j)
    run_block(s, pos3, feat3, 512, 128, 16, idx3, enc[2].blocks[j], bf);

  // enc4 td: 512 -> 128, din=131 -> 256
  k_fps<<<dim3(2), dim3(256), 0, s>>>(pos3, 512, 128, sidx, fpsd);
  LAUNCH1D(k_gatherpos, 2*128*3, pos3, sidx, 512, 128, pos4, 2*128*3);
  k_knn<16><<<dim3(CEILDIV(128,128), 2), dim3(128), 0, s>>>(pos4, 128, pos3, 512, tdidx, nullptr);
  LAUNCH1D(k_tdg, 2*128*16*131, pos3, pos4, feat3, tdidx, 128, 512, 16, 128, gbuf, 2*128*16*131);
  gemm(s, gbuf, enc[3].td_lin.w, nullptr, enc[3].td_bn.g, enc[3].td_bn.be,
       nullptr, 1, ybuf, 2*128*16, 131, 256);
  LAUNCH1D(k_maxns, 2*128*256, ybuf, 16, 256, feat4, 2*128*256);
  k_knn<16><<<dim3(CEILDIV(128,128), 2), dim3(128), 0, s>>>(pos4, 128, pos4, 128, idx4, nullptr);
  for (int j = 0; j < enc[3].nb; ++j)
    run_block(s, pos4, feat4, 128, 256, 16, idx4, enc[3].blocks[j], bf);

  // ======================== DECODER ========================
  // dec4: tu_head + 1 block
  LAUNCH1D(k_mean, 2*256, feat4, 128, 256, meanb, 2*256);
  gemm(s, meanb, dec[3].l2.w, dec[3].l2.b, nullptr, nullptr, nullptr, 1, gvec, 2, 256, 256);
  LAUNCH1D(k_catb, 2*128*512, feat4, 128, 256, gvec, gbuf, 2*128*512);
  gemm(s, gbuf, dec[3].l1.w, dec[3].l1.b, dec[3].bn1.g, dec[3].bn1.be,
       nullptr, 1, feat4, 2*128, 512, 256);
  run_block(s, pos4, feat4, 128, 256, 16, idx4, dec[3].blk, bf);

  // dec3: fine L3 (512,128) <- coarse L4
  gemm(s, feat3, dec[2].l1.w, dec[2].l1.b, dec[2].bn1.g, dec[2].bn1.be, nullptr, 1,
       bf.ty, 2*512, 128, 128);
  gemm(s, feat4, dec[2].l2.w, dec[2].l2.b, dec[2].bn2.g, dec[2].bn2.be, nullptr, 1,
       bf.ty2, 2*128, 256, 128);
  k_knn<3><<<dim3(CEILDIV(512,128), 2), dim3(128), 0, s>>>(pos3, 512, pos4, 128, iidx, idist);
  LAUNCH1D(k_interp, 2*512*128, bf.ty, bf.ty2, iidx, idist, 512, 128, 128, feat3, 2*512*128);
  run_block(s, pos3, feat3, 512, 128, 16, idx3, dec[2].blk, bf);

  // dec2: fine L2 (2048,64) <- coarse L3
  gemm(s, feat2, dec[1].l1.w, dec[1].l1.b, dec[1].bn1.g, dec[1].bn1.be, nullptr, 1,
       bf.ty, 2*2048, 64, 64);
  gemm(s, feat3, dec[1].l2.w, dec[1].l2.b, dec[1].bn2.g, dec[1].bn2.be, nullptr, 1,
       bf.ty2, 2*512, 128, 64);
  k_knn<3><<<dim3(CEILDIV(2048,128), 2), dim3(128), 0, s>>>(pos2, 2048, pos3, 512, iidx, idist);
  LAUNCH1D(k_interp, 2*2048*64, bf.ty, bf.ty2, iidx, idist, 2048, 512, 64, feat2, 2*2048*64);
  run_block(s, pos2, feat2, 2048, 64, 16, idx2, dec[1].blk, bf);

  // dec1: fine L1 (8192,32) <- coarse L2
  gemm(s, feat1, dec[0].l1.w, dec[0].l1.b, dec[0].bn1.g, dec[0].bn1.be, nullptr, 1,
       bf.ty, 2*8192, 32, 32);
  gemm(s, feat2, dec[0].l2.w, dec[0].l2.b, dec[0].bn2.g, dec[0].bn2.be, nullptr, 1,
       bf.ty2, 2*2048, 64, 32);
  k_knn<3><<<dim3(CEILDIV(8192,128), 2), dim3(128), 0, s>>>(P, 8192, pos2, 2048, iidx, idist);
  LAUNCH1D(k_interp, 2*8192*32, bf.ty, bf.ty2, iidx, idist, 8192, 2048, 32, feat1, 2*8192*32);
  run_block(s, P, feat1, 8192, 32, 8, idx1, dec[0].blk, bf);

  // ======================== OUTPUTS (x4, x3, x2, x1) ========================
  float* OUT = (float*)d_out;
  hipMemcpyAsync(OUT,                       feat4, sizeof(float)*2*128*256,  hipMemcpyDeviceToDevice, s);
  hipMemcpyAsync(OUT + 65536,               feat3, sizeof(float)*2*512*128,  hipMemcpyDeviceToDevice, s);
  hipMemcpyAsync(OUT + 65536 + 131072,      feat2, sizeof(float)*2*2048*64,  hipMemcpyDeviceToDevice, s);
  hipMemcpyAsync(OUT + 65536 + 131072 + 262144, feat1, sizeof(float)*2*8192*32, hipMemcpyDeviceToDevice, s);
}
